// DTIPredictor_81990925681126
// MI455X (gfx1250) — matrix operand order repaired and checked by reading, not compiled, not run on hardware
//
#include <hip/hip_runtime.h>
#include <hip/hip_bf16.h>
#include <math.h>

// ---------------------------------------------------------------------------
// DTI predictor forward, MI455X (gfx1250, wave32).
// All dense layers run through one bf16 WMMA GEMM kernel
// (v_wmma_f32_16x16x32_bf16, fp32 accumulate) with fused BN/ReLU/GELU.
// Weights are transpose-converted once to bf16 [N,K] so both LDS staging and
// per-fragment reads are wide contiguous (b128) accesses.
// ---------------------------------------------------------------------------

#define BNEPS 1e-5f

typedef __attribute__((ext_vector_type(16))) __bf16          v16bf;
typedef __attribute__((ext_vector_type(8)))  unsigned short  v8us;
typedef __attribute__((ext_vector_type(8)))  float           v8f;

struct U16x16 { v8us lo, hi; };   // 32 bytes, bit_cast-compatible with v16bf

__device__ __forceinline__ unsigned short f2bf(float f) {
  // round-to-nearest-even fp32 -> bf16
  unsigned int u = __builtin_bit_cast(unsigned int, f);
  u += 0x7FFFu + ((u >> 16) & 1u);
  return (unsigned short)(u >> 16);
}

// ---------------------------------------------------------------------------
// Tiled transpose + fp32->bf16 convert: W[K,N] -> WT[N,K] (bf16)
// ---------------------------------------------------------------------------
__global__ __launch_bounds__(256)
void convert_wT_kernel(const float* __restrict__ W, unsigned short* __restrict__ WT,
                       int K, int N)
{
  __shared__ float tile[32][33];
  const int tx = threadIdx.x;          // 32
  const int ty = threadIdx.y;          // 8
  const int n0 = blockIdx.x * 32;
  const int k0 = blockIdx.y * 32;
#pragma unroll
  for (int r = 0; r < 4; ++r)
    tile[ty + 8 * r][tx] = W[(size_t)(k0 + ty + 8 * r) * N + n0 + tx];
  __syncthreads();
#pragma unroll
  for (int r = 0; r < 4; ++r)
    WT[(size_t)(n0 + ty + 8 * r) * K + k0 + tx] = f2bf(tile[tx][ty + 8 * r]);
}

// ---------------------------------------------------------------------------
// Generic GEMM:  C[M,N] = epi( preA(A)[M,K] x W[K,N] + bias )
//   PRE : BN(K-indexed)+ReLU applied while staging A (scale/shift pre-folded
//         into LDS once; only used with K <= 128)
//   EPI : 1 = +bias ; 2 = +bias, BN(N-indexed), ReLU ; 3 = +bias, exact GELU
// Tile 128x128, 256 threads = 8 waves; wave w computes rows [16w,16w+16),
// 8 fragments across N. K stepped by 32 through bf16 LDS tiles.
// WT is the pre-converted bf16 weight, transposed to [N,K].
// ---------------------------------------------------------------------------
template <bool PRE, int EPI>
__global__ __launch_bounds__(256)
void gemm_bf16_epi(const float* __restrict__ A, const unsigned short* __restrict__ WT,
                   const float* __restrict__ bias,
                   const float* __restrict__ pg, const float* __restrict__ pb,
                   const float* __restrict__ pm, const float* __restrict__ pv,
                   const float* __restrict__ qg, const float* __restrict__ qb,
                   const float* __restrict__ qm, const float* __restrict__ qv,
                   float* __restrict__ C, int M, int N, int K)
{
  __shared__ unsigned short lds_a[128][32];    // A tile, [m][k]
  __shared__ unsigned short lds_bT[128][32];   // W tile, [n][k]
  __shared__ float s_scale[128];
  __shared__ float s_shift[128];

  const int t    = threadIdx.x;
  const int wid  = t >> 5;
  const int lane = t & 31;
  const int lrow = lane & 15;
  const int half = lane >> 4;

  const int blockM = blockIdx.x * 128;
  const int blockN = blockIdx.y * 128;

  if (PRE) {                                   // fold BN into scale/shift once
    for (int k = t; k < K && k < 128; k += 256) {
      const float sc = pg[k] * rsqrtf(pv[k] + BNEPS);
      s_scale[k] = sc;
      s_shift[k] = pb[k] - pm[k] * sc;
    }
    __syncthreads();
  }

  v8f acc[8];
#pragma unroll
  for (int i = 0; i < 8; ++i)
#pragma unroll
    for (int j = 0; j < 8; ++j) acc[i][j] = 0.0f;

  for (int k0 = 0; k0 < K; k0 += 32) {
    // --- stage A tile (128x32): 2 threads/row, float4 loads, bf16 stores ---
    {
      const int row = t >> 1;
      const int cb  = (t & 1) * 16;
      const float4* ap = (const float4*)(A + (size_t)(blockM + row) * K + k0 + cb);
      v8us lo, hi;
#pragma unroll
      for (int q = 0; q < 4; ++q) {
        const float4 v = ap[q];
        float f0 = v.x, f1 = v.y, f2 = v.z, f3 = v.w;
        if (PRE) {
          const int kg = k0 + cb + q * 4;
          f0 = fmaxf(f0 * s_scale[kg + 0] + s_shift[kg + 0], 0.0f);
          f1 = fmaxf(f1 * s_scale[kg + 1] + s_shift[kg + 1], 0.0f);
          f2 = fmaxf(f2 * s_scale[kg + 2] + s_shift[kg + 2], 0.0f);
          f3 = fmaxf(f3 * s_scale[kg + 3] + s_shift[kg + 3], 0.0f);
        }
        if (q < 2) {
          lo[q * 4 + 0] = f2bf(f0); lo[q * 4 + 1] = f2bf(f1);
          lo[q * 4 + 2] = f2bf(f2); lo[q * 4 + 3] = f2bf(f3);
        } else {
          hi[(q - 2) * 4 + 0] = f2bf(f0); hi[(q - 2) * 4 + 1] = f2bf(f1);
          hi[(q - 2) * 4 + 2] = f2bf(f2); hi[(q - 2) * 4 + 3] = f2bf(f3);
        }
      }
      *(v8us*)&lds_a[row][cb]     = lo;
      *(v8us*)&lds_a[row][cb + 8] = hi;
    }
    // --- stage W tile (128n x 32k) from bf16 WT[N,K]: pure wide copies ---
    {
      const int n  = t >> 1;
      const int kb = (t & 1) * 16;
      const v8us* wp = (const v8us*)(WT + (size_t)(blockN + n) * K + k0 + kb);
      *(v8us*)&lds_bT[n][kb]     = wp[0];
      *(v8us*)&lds_bT[n][kb + 8] = wp[1];
    }
    __syncthreads();

    // A fragment: lanes 0-15 hold K 0-15, lanes 16-31 hold K 16-31
    U16x16 apack;
    {
      const int m  = wid * 16 + lrow;
      const int kb = half * 16;
      apack.lo = *(const v8us*)&lds_a[m][kb];
      apack.hi = *(const v8us*)&lds_a[m][kb + 8];
    }
    const v16bf af = __builtin_bit_cast(v16bf, apack);

#pragma unroll
    for (int nf = 0; nf < 8; ++nf) {
      // B fragment: lanes 0-15 hold K 0-15, lanes 16-31 hold K 16-31
      const int n  = nf * 16 + lrow;
      const int kb = half * 16;
      U16x16 bpack;
      bpack.lo = *(const v8us*)&lds_bT[n][kb];
      bpack.hi = *(const v8us*)&lds_bT[n][kb + 8];
      const v16bf bfr = __builtin_bit_cast(v16bf, bpack);
      acc[nf] = __builtin_amdgcn_wmma_f32_16x16x32_bf16(
          false, af, false, bfr, (short)0, acc[nf], false, false);
    }
    __syncthreads();
  }

  // --- epilogue: D layout VGPR r -> M=r (lanes 0-15) / M=r+8 (lanes 16-31) ---
#pragma unroll
  for (int nf = 0; nf < 8; ++nf) {
    const int col = blockN + nf * 16 + lrow;
    float qsc = 0.0f, qsh = 0.0f, bv = bias[col];
    if (EPI == 2) {
      qsc = qg[col] * rsqrtf(qv[col] + BNEPS);
      qsh = qb[col] - qm[col] * qsc;
    }
#pragma unroll
    for (int r = 0; r < 8; ++r) {
      const int row = blockM + wid * 16 + half * 8 + r;
      float v = acc[nf][r] + bv;
      if (EPI == 2) {
        v = fmaxf(v * qsc + qsh, 0.0f);
      } else if (EPI == 3) {
        v = 0.5f * v * (1.0f + erff(v * 0.70710678118654752f));
      }
      C[(size_t)row * N + col] = v;
    }
  }
}

// ---------------------------------------------------------------------------
// Edge gate: sigmoid(relu(ea @ w1 + b1) @ w2 + b2), w1:[FIN,32], w2:[32,1]
// ---------------------------------------------------------------------------
template <int FIN>
__global__ void edge_gate_kernel(const float* __restrict__ ea,
                                 const float* __restrict__ w1, const float* __restrict__ b1,
                                 const float* __restrict__ w2, const float* __restrict__ b2,
                                 float* __restrict__ gate, int E)
{
  const int e = blockIdx.x * blockDim.x + threadIdx.x;
  if (e >= E) return;
  float in[FIN];
#pragma unroll
  for (int i = 0; i < FIN; ++i) in[i] = ea[(size_t)e * FIN + i];
  float acc = b2[0];
#pragma unroll 8
  for (int j = 0; j < 32; ++j) {
    float h = b1[j];
#pragma unroll
    for (int i = 0; i < FIN; ++i) h += in[i] * w1[i * 32 + j];
    acc += fmaxf(h, 0.0f) * w2[j];
  }
  gate[e] = 1.0f / (1.0f + __expf(-acc));
}

__global__ void fill0_kernel(float* __restrict__ p, size_t n)
{
  size_t i = (size_t)blockIdx.x * blockDim.x + threadIdx.x;
  if (i < n) p[i] = 0.0f;
}

// msg = gate[e] * x[src[e]], scatter-add at dst[e].  32 lanes per edge,
// float4 per lane over H=128; f32 global atomics.
__global__ void scatter_kernel(const float* __restrict__ x,
                               const int* __restrict__ src, const int* __restrict__ dst,
                               const float* __restrict__ gate,
                               float* __restrict__ agg, int E)
{
  const int lane = threadIdx.x & 31;
  const int e = blockIdx.x * (blockDim.x >> 5) + (threadIdx.x >> 5);
  if (e >= E) return;
  const int   s = src[e];
  const int   d = dst[e];
  const float g = gate[e];
  const float4 v = ((const float4*)(x + (size_t)s * 128))[lane];
  float* out = agg + (size_t)d * 128 + lane * 4;
  atomicAdd(out + 0, g * v.x);
  atomicAdd(out + 1, g * v.y);
  atomicAdd(out + 2, g * v.z);
  atomicAdd(out + 3, g * v.w);
}

// segment_max over contiguous per-graph node ranges (batch = arange // per)
__global__ void pool_max_kernel(const float* __restrict__ x, float* __restrict__ out,
                                int per, int ldo, int off)
{
  const int b = blockIdx.x;
  const int f = threadIdx.x;             // 128 features
  const float* p = x + (size_t)b * per * 128 + f;
  float m = p[0];
  for (int i = 1; i < per; ++i) m = fmaxf(m, p[(size_t)i * 128]);
  out[(size_t)b * ldo + off + f] = m;
}

// torch GRUCell (r,z,n ordering); input = drug pooled, hidden = fp MLP out
__global__ void gru_kernel(const float* __restrict__ xin, const float* __restrict__ hin,
                           const float* __restrict__ wi, const float* __restrict__ bi,
                           const float* __restrict__ wh, const float* __restrict__ bh,
                           float* __restrict__ out, int ldo)
{
  const int idx = blockIdx.x * blockDim.x + threadIdx.x;   // B*128
  const int b = idx >> 7, j = idx & 127;
  const float* xb = xin + (size_t)b * 128;
  const float* hb = hin + (size_t)b * 128;
  float ir = bi[j], iz = bi[128 + j], in_ = bi[256 + j];
  float hr = bh[j], hz = bh[128 + j], hn  = bh[256 + j];
  for (int k = 0; k < 128; ++k) {
    const float xv = xb[k], hv = hb[k];
    ir  += xv * wi[(size_t)j * 128 + k];
    iz  += xv * wi[(size_t)(128 + j) * 128 + k];
    in_ += xv * wi[(size_t)(256 + j) * 128 + k];
    hr  += hv * wh[(size_t)j * 128 + k];
    hz  += hv * wh[(size_t)(128 + j) * 128 + k];
    hn  += hv * wh[(size_t)(256 + j) * 128 + k];
  }
  const float r = 1.0f / (1.0f + __expf(-(ir + hr)));
  const float z = 1.0f / (1.0f + __expf(-(iz + hz)));
  const float n = tanhf(in_ + r * hn);
  out[(size_t)b * ldo + j] = (1.0f - z) * n + z * hb[j];
}

__global__ void head_l3_kernel(const float* __restrict__ h2, const float* __restrict__ w,
                               const float* __restrict__ b, float* __restrict__ out)
{
  const int i = blockIdx.x * blockDim.x + threadIdx.x;     // 256 rows
  if (i >= 256) return;
  float acc = b[0];
  for (int k = 0; k < 512; ++k) acc += h2[(size_t)i * 512 + k] * w[k];
  out[i] = acc;
}

// ---------------------------------------------------------------------------
// Orchestration
// ---------------------------------------------------------------------------
extern "C" void kernel_launch(void* const* d_in, const int* in_sizes, int n_in,
                              void* d_out, int out_size, void* d_ws, size_t ws_size,
                              hipStream_t stream)
{
  (void)in_sizes; (void)n_in; (void)out_size;
  const int NMOL = 10240, EMOL = 40960, NPROT = 102400, EPROT = 1228800;

  // ---- inputs (setup_inputs insertion order) ----
  const float* drug_x  = (const float*)d_in[0];
  const int*   mol_ei  = (const int*)  d_in[1];
  const float* mol_ea  = (const float*)d_in[2];
  const float* fp_in   = (const float*)d_in[4];
  const float* prot_x  = (const float*)d_in[5];
  const int*   prot_ei = (const int*)  d_in[6];
  const float* prot_ea = (const float*)d_in[7];

  // ---- params: jax pytree flatten (dict keys sorted, tuples in order) ----
  // 9-20: fp{bn1(b,g,m,v),bn2(b,g,m,v),l1(b,w),l2(b,w)}
  // 21-24: gru{bh,bi,wh,wi}   25-30: mlp{l1(b,w),l2(b,w),l3(b,w)}
  // 31-34: mol_edge{l1(b,w),l2(b,w)}  35-36: mol_fc{b,w}
  // 37-66: mol_gnn x3 {bn1(b,g,m,v),bn2(b,g,m,v),lin(b,w)}
  // 67-70: prot_edge  71-72: prot_fc{b,w}  73-102: prot_gnn x3
  auto P = [&](int i) { return (const float*)d_in[i]; };

  // ---- workspace layout ----
  float* wsp = (float*)d_ws;
  size_t off = 0;
  auto alloc = [&](size_t n) { float* p = wsp + off; off += n; return p; };
  float* Xp  = alloc((size_t)NPROT * 128);   // prot node features
  float* Ap  = alloc((size_t)NPROT * 128);   // prot aggregation
  float* Xm  = alloc((size_t)NMOL * 128);
  float* Am  = alloc((size_t)NMOL * 128);
  float* Gp  = alloc(EPROT);
  float* Gm  = alloc(EMOL);
  float* FH1 = alloc(256 * 128);
  float* FPH = alloc(256 * 128);
  float* CAT = alloc(256 * 256);             // [gru_out | prot_pool]
  float* DP  = alloc(256 * 128);             // drug pooled (GRU input)
  float* Hh1 = alloc(256 * 2048);
  float* Hh2 = alloc(256 * 512);
  unsigned short* WT = (unsigned short*)alloc(524288);  // bf16 WT scratch, 1M elems
  if (ws_size < off * sizeof(float)) return; // workspace too small: bail

  // convert W[K,N] fp32 -> WT[N,K] bf16, then run templated GEMM
  auto gemm = [&](const float* A, const float* Wf, const float* bias,
                  const float* pg, const float* pb, const float* pm, const float* pv,
                  const float* qg, const float* qb, const float* qm, const float* qv,
                  float* C, int M, int N, int K, int epi, bool pre) {
    convert_wT_kernel<<<dim3(N / 32, K / 32), dim3(32, 8), 0, stream>>>(Wf, WT, K, N);
    dim3 grid(M / 128, N / 128);
    if (pre)
      gemm_bf16_epi<true, 2><<<grid, 256, 0, stream>>>(A, WT, bias, pg, pb, pm, pv,
                                                       qg, qb, qm, qv, C, M, N, K);
    else if (epi == 1)
      gemm_bf16_epi<false, 1><<<grid, 256, 0, stream>>>(A, WT, bias, pg, pb, pm, pv,
                                                        qg, qb, qm, qv, C, M, N, K);
    else if (epi == 2)
      gemm_bf16_epi<false, 2><<<grid, 256, 0, stream>>>(A, WT, bias, pg, pb, pm, pv,
                                                        qg, qb, qm, qv, C, M, N, K);
    else
      gemm_bf16_epi<false, 3><<<grid, 256, 0, stream>>>(A, WT, bias, pg, pb, pm, pv,
                                                        qg, qb, qm, qv, C, M, N, K);
  };
  const float* Z = nullptr;

  // ================= protein branch =================
  gemm(prot_x, P(72), P(71), Z,Z,Z,Z, Z,Z,Z,Z, Xp, NPROT, 128, 1280, 1, false);
  edge_gate_kernel<2><<<(EPROT + 255) / 256, 256, 0, stream>>>(
      prot_ea, P(68), P(67), P(70), P(69), Gp, EPROT);
  for (int l = 0; l < 3; ++l) {
    const int q = 73 + 10 * l;  // bn1.b,g,m,v | bn2.b,g,m,v | lin.b | lin.w
    fill0_kernel<<<(int)(((size_t)NPROT * 128 + 255) / 256), 256, 0, stream>>>(
        Ap, (size_t)NPROT * 128);
    scatter_kernel<<<(EPROT + 7) / 8, 256, 0, stream>>>(
        Xp, prot_ei, prot_ei + EPROT, Gp, Ap, EPROT);
    gemm(Ap, P(q + 9), P(q + 8),
         P(q + 1), P(q + 0), P(q + 2), P(q + 3),     // pre: bn1 (g,b,m,v)
         P(q + 5), P(q + 4), P(q + 6), P(q + 7),     // post: bn2
         Xp, NPROT, 128, 128, 2, true);
  }
  pool_max_kernel<<<256, 128, 0, stream>>>(Xp, CAT, 400, 256, 128);

  // ================= drug branch =================
  gemm(drug_x, P(36), P(35), Z,Z,Z,Z, Z,Z,Z,Z, Xm, NMOL, 128, 64, 1, false);
  edge_gate_kernel<3><<<(EMOL + 255) / 256, 256, 0, stream>>>(
      mol_ea, P(32), P(31), P(34), P(33), Gm, EMOL);
  for (int l = 0; l < 3; ++l) {
    const int q = 37 + 10 * l;
    fill0_kernel<<<(int)(((size_t)NMOL * 128 + 255) / 256), 256, 0, stream>>>(
        Am, (size_t)NMOL * 128);
    scatter_kernel<<<(EMOL + 7) / 8, 256, 0, stream>>>(
        Xm, mol_ei, mol_ei + EMOL, Gm, Am, EMOL);
    gemm(Am, P(q + 9), P(q + 8),
         P(q + 1), P(q + 0), P(q + 2), P(q + 3),
         P(q + 5), P(q + 4), P(q + 6), P(q + 7),
         Xm, NMOL, 128, 128, 2, true);
  }
  pool_max_kernel<<<256, 128, 0, stream>>>(Xm, DP, 40, 128, 0);

  // ================= fingerprint MLP =================
  gemm(fp_in, P(18), P(17), Z,Z,Z,Z, P(10), P(9),  P(11), P(12), FH1, 256, 128, 1024, 2, false);
  gemm(FH1,   P(20), P(19), Z,Z,Z,Z, P(14), P(13), P(15), P(16), FPH, 256, 128, 128,  2, false);

  // ================= GRU (writes concat cols 0..127) =================
  gru_kernel<<<(256 * 128) / 256, 256, 0, stream>>>(
      DP, FPH, P(24), P(22), P(23), P(21), CAT, 256);

  // ================= head MLP =================
  gemm(CAT, P(26), P(25), Z,Z,Z,Z, Z,Z,Z,Z, Hh1, 256, 2048, 256,  3, false);
  gemm(Hh1, P(28), P(27), Z,Z,Z,Z, Z,Z,Z,Z, Hh2, 256, 512,  2048, 3, false);
  head_l3_kernel<<<1, 256, 0, stream>>>(Hh2, P(30), P(29), (float*)d_out);
}
